// GCNNodeClassifier_18107582119956
// MI455X (gfx1250) — matrix operand order repaired and checked by reading, not compiled, not run on hardware
//
#include <hip/hip_runtime.h>

// ---------------------------------------------------------------------------
// Types for CDNA5 WMMA
// ---------------------------------------------------------------------------
typedef __attribute__((ext_vector_type(16))) __bf16       v16bf;
typedef __attribute__((ext_vector_type(8)))  float        v8f;
typedef __attribute__((ext_vector_type(4)))  unsigned int v4u;

__device__ __forceinline__ unsigned short f2bf(float f) {
    unsigned int u = __float_as_uint(f);
    u += 0x7FFFu + ((u >> 16) & 1u);   // round-to-nearest-even
    return (unsigned short)(u >> 16);
}

// ---------------------------------------------------------------------------
// Utility kernels
// ---------------------------------------------------------------------------
__global__ void zero_u32_kernel(unsigned int* __restrict__ p, long long n) {
    long long i = blockIdx.x * (long long)blockDim.x + threadIdx.x;
    if (i < n) p[i] = 0u;
}

__global__ void f32_to_bf16_kernel(const float* __restrict__ in,
                                   unsigned short* __restrict__ out, long long n) {
    long long i = blockIdx.x * (long long)blockDim.x + threadIdx.x;
    if (i < n) out[i] = f2bf(in[i]);
}

// Degree histogram over destination nodes (self-loop added later as +1).
__global__ void degree_kernel(const long long* __restrict__ col,
                              unsigned int* __restrict__ deg, int E) {
    int e = blockIdx.x * blockDim.x + threadIdx.x;
    if (e < E) atomicAdd(&deg[(int)col[e]], 1u);
}

__global__ void dinv_kernel(const unsigned int* __restrict__ deg,
                            float* __restrict__ dinv, int N) {
    int i = blockIdx.x * blockDim.x + threadIdx.x;
    if (i < N) dinv[i] = rsqrtf((float)deg[i] + 1.0f);   // +1 = self-loop
}

__global__ void norm_kernel(const long long* __restrict__ row,
                            const long long* __restrict__ col,
                            const float* __restrict__ dinv,
                            float* __restrict__ nrm, int E) {
    int e = blockIdx.x * blockDim.x + threadIdx.x;
    if (e < E) nrm[e] = dinv[(int)row[e]] * dinv[(int)col[e]];
}

// Pack W (K x COUT, row-major f32) into CDNA5 WMMA B-matrix layout (bf16):
// out[((nt*KT + kt)*32 + lane)*16 + kk]  holds  W[kt*32 + (lane<16?0:8) + (kk<8?kk:kk+8)][nt*16 + lane%16]
__global__ void pack_w_kernel(const float* __restrict__ W,
                              unsigned short* __restrict__ out, int K, int COUT) {
    int t = blockIdx.x * blockDim.x + threadIdx.x;
    int total = K * COUT;
    if (t >= total) return;
    int ktiles = K >> 5;
    int kk   = t & 15;
    int lane = (t >> 4) & 31;
    int kt   = (t >> 9) % ktiles;
    int nt   = t / (ktiles << 9);
    int n = nt * 16 + (lane & 15);
    int k = kt * 32 + ((lane < 16) ? 0 : 8) + ((kk < 8) ? kk : kk + 8);
    out[t] = f2bf(W[k * COUT + n]);
}

// ---------------------------------------------------------------------------
// GEMM: H = A(bf16, N x CIN row-major) * Bpacked  -> Hout (f32, N x COUT)
// One wave per 16x16 output tile; K-loop of v_wmma_f32_16x16x32_bf16.
// ---------------------------------------------------------------------------
template <int CIN, int COUT>
__global__ void gemm_wmma_kernel(const unsigned short* __restrict__ Abf,
                                 const unsigned short* __restrict__ Bpk,
                                 float* __restrict__ Hout, int nRowTiles) {
    constexpr int KT = CIN / 32;
    constexpr int NT = COUT / 16;
    int wid  = threadIdx.x >> 5;
    int lane = threadIdx.x & 31;
    int tile = blockIdx.x * (blockDim.x >> 5) + wid;
    int totalTiles = nRowTiles * NT;
    if (tile >= totalTiles) return;
    int mt = tile / NT;
    int nt = tile % NT;

    // A layout (16-bit 16x32): lane holds row M = lane%16;
    // lanes 0-15: K in {0..7, 16..23}; lanes 16-31: K in {8..15, 24..31}.
    int row = mt * 16 + (lane & 15);
    const unsigned short* arow = Abf + (size_t)row * CIN;
    int aoff = (lane < 16) ? 0 : 8;
    // B packed: 16 contiguous K-values per lane per k-tile.
    const unsigned short* bbase = Bpk + ((size_t)(nt * KT) * 32 + (size_t)lane) * 16;

    v8f acc = {};
#pragma unroll
    for (int kt = 0; kt < KT; ++kt) {
        union { v4u u[2]; v16bf v; } A;
        A.u[0] = *(const v4u*)(arow + kt * 32 + aoff);
        A.u[1] = *(const v4u*)(arow + kt * 32 + aoff + 16);
        union { v4u u[2]; v16bf v; } Bm;
        const unsigned short* bp = bbase + (size_t)kt * 512;
        Bm.u[0] = *(const v4u*)(bp);
        Bm.u[1] = *(const v4u*)(bp + 8);
        acc = __builtin_amdgcn_wmma_f32_16x16x32_bf16(
            /*neg_a=*/false, A.v, /*neg_b=*/false, Bm.v,
            /*c_mod=*/(short)0, acc, /*reuse_a=*/false, /*reuse_b=*/false);
    }

    // D layout: VGPR i -> row (i + (lane<16?0:8)), lane%16 -> column.
    int mbase = mt * 16 + ((lane < 16) ? 0 : 8);
    float* orow = Hout + (size_t)mbase * COUT + nt * 16 + (lane & 15);
#pragma unroll
    for (int i = 0; i < 8; ++i)
        orow[(size_t)i * COUT] = acc[i];
}

// ---------------------------------------------------------------------------
// Scatter-add: acc[col[e]] += h[row[e]] * norm[e]   (C/4 lanes per edge, float4)
// ---------------------------------------------------------------------------
template <int C>
__global__ void scatter_kernel(const float* __restrict__ h,
                               const long long* __restrict__ row,
                               const long long* __restrict__ col,
                               const float* __restrict__ nrm,
                               float* __restrict__ acc, int E) {
    constexpr int TPE = C / 4;
    long long g = blockIdx.x * (long long)blockDim.x + threadIdx.x;
    if (g >= (long long)E * TPE) return;
    int e = (int)(g / TPE);
    int q = (int)(g % TPE);
    int r = (int)row[e];
    int c = (int)col[e];
    float w = nrm[e];
    const float4 hv = *(const float4*)(h + (size_t)r * C + q * 4);
    float* dst = acc + (size_t)c * C + q * 4;
    atomicAdd(dst + 0, hv.x * w);
    atomicAdd(dst + 1, hv.y * w);
    atomicAdd(dst + 2, hv.z * w);
    atomicAdd(dst + 3, hv.w * w);
}

// ---------------------------------------------------------------------------
// Post: out = acc + h*dinv^2 (self-loop) + bias; optional ReLU; bf16 or f32 out
// ---------------------------------------------------------------------------
template <int C, bool RELU, bool BF16OUT>
__global__ void post_kernel(const float* __restrict__ acc, const float* __restrict__ h,
                            const float* __restrict__ dinv, const float* __restrict__ bias,
                            unsigned short* __restrict__ obf, float* __restrict__ of32, int N) {
    long long g = blockIdx.x * (long long)blockDim.x + threadIdx.x;
    if (g >= (long long)N * C) return;
    int i = (int)(g / C);
    int c = (int)(g % C);
    float dv = dinv[i];
    float v = acc[g] + h[g] * dv * dv + bias[c];
    if (RELU) v = fmaxf(v, 0.0f);
    if (BF16OUT) obf[g] = f2bf(v);
    else         of32[g] = v;
}

// ---------------------------------------------------------------------------
// Launcher
// ---------------------------------------------------------------------------
extern "C" void kernel_launch(void* const* d_in, const int* in_sizes, int n_in,
                              void* d_out, int out_size, void* d_ws, size_t ws_size,
                              hipStream_t stream) {
    (void)in_sizes; (void)n_in; (void)out_size; (void)ws_size;

    const int N = 50000, E = 800000;
    const float*      x   = (const float*)d_in[0];
    const long long*  ei  = (const long long*)d_in[1];
    const float*      W1  = (const float*)d_in[2];
    const float*      b1  = (const float*)d_in[3];
    const float*      W2  = (const float*)d_in[4];
    const float*      b2  = (const float*)d_in[5];
    const float*      W3  = (const float*)d_in[6];
    const float*      b3  = (const float*)d_in[7];
    float*            out = (float*)d_out;

    const long long* row = ei;
    const long long* col = ei + E;

    // Workspace carve-out (256B aligned slices).
    char* ws = (char*)d_ws;
    size_t o = 0;
    auto take = [&](size_t b) { void* p = ws + o; o += (b + 255) & ~(size_t)255; return p; };
    unsigned short* xbf  = (unsigned short*)take((size_t)N * 128 * 2);
    unsigned short* abf  = (unsigned short*)take((size_t)N * 128 * 2);
    float*          h    = (float*)take((size_t)N * 128 * 4);
    float*          acc  = (float*)take((size_t)N * 128 * 4);
    unsigned int*   deg  = (unsigned int*)take((size_t)N * 4);
    float*          dinv = (float*)take((size_t)N * 4);
    float*          nrm  = (float*)take((size_t)E * 4);
    unsigned short* wp1  = (unsigned short*)take(128 * 128 * 2);
    unsigned short* wp2  = (unsigned short*)take(128 * 128 * 2);
    unsigned short* wp3  = (unsigned short*)take(128 * 64 * 2);

    dim3 B(256);
    auto blocks = [](long long n) { return dim3((unsigned)((n + 255) / 256)); };

    // --- shared precompute -------------------------------------------------
    zero_u32_kernel<<<blocks(N), B, 0, stream>>>(deg, N);
    degree_kernel<<<blocks(E), B, 0, stream>>>(col, deg, E);
    dinv_kernel<<<blocks(N), B, 0, stream>>>(deg, dinv, N);
    norm_kernel<<<blocks(E), B, 0, stream>>>(row, col, dinv, nrm, E);
    f32_to_bf16_kernel<<<blocks((long long)N * 128), B, 0, stream>>>(x, xbf, (long long)N * 128);
    pack_w_kernel<<<blocks(128 * 128), B, 0, stream>>>(W1, wp1, 128, 128);
    pack_w_kernel<<<blocks(128 * 128), B, 0, stream>>>(W2, wp2, 128, 128);
    pack_w_kernel<<<blocks(128 * 64),  B, 0, stream>>>(W3, wp3, 128, 64);

    const int rowTiles = N / 16;  // 3125, exact

    // --- layer 1: 128 -> 128, ReLU ----------------------------------------
    gemm_wmma_kernel<128, 128><<<dim3((unsigned)((rowTiles * 8 + 7) / 8)), B, 0, stream>>>(xbf, wp1, h, rowTiles);
    zero_u32_kernel<<<blocks((long long)N * 128), B, 0, stream>>>((unsigned int*)acc, (long long)N * 128);
    scatter_kernel<128><<<blocks((long long)E * 32), B, 0, stream>>>(h, row, col, nrm, acc, E);
    post_kernel<128, true, true><<<blocks((long long)N * 128), B, 0, stream>>>(acc, h, dinv, b1, abf, nullptr, N);

    // --- layer 2: 128 -> 128, ReLU ----------------------------------------
    gemm_wmma_kernel<128, 128><<<dim3((unsigned)((rowTiles * 8 + 7) / 8)), B, 0, stream>>>(abf, wp2, h, rowTiles);
    zero_u32_kernel<<<blocks((long long)N * 128), B, 0, stream>>>((unsigned int*)acc, (long long)N * 128);
    scatter_kernel<128><<<blocks((long long)E * 32), B, 0, stream>>>(h, row, col, nrm, acc, E);
    post_kernel<128, true, true><<<blocks((long long)N * 128), B, 0, stream>>>(acc, h, dinv, b2, abf, nullptr, N);

    // --- layer 3: 128 -> 64, no ReLU, f32 out ------------------------------
    gemm_wmma_kernel<128, 64><<<dim3((unsigned)((rowTiles * 4 + 7) / 8)), B, 0, stream>>>(abf, wp3, h, rowTiles);
    zero_u32_kernel<<<blocks((long long)N * 64), B, 0, stream>>>((unsigned int*)acc, (long long)N * 64);
    scatter_kernel<64><<<blocks((long long)E * 16), B, 0, stream>>>(h, row, col, nrm, acc, E);
    post_kernel<64, false, false><<<blocks((long long)N * 64), B, 0, stream>>>(acc, h, dinv, b3, nullptr, out, N);
}
